// targetPredict_59820304498760
// MI455X (gfx1250) — matrix operand order repaired and checked by reading, not compiled, not run on hardware
//
#include <hip/hip_runtime.h>
#include <hip/hip_bf16.h>
#include <cmath>

typedef _Float16 v16h __attribute__((ext_vector_type(16)));
typedef float    v8f  __attribute__((ext_vector_type(8)));

#define NCLS   21
#define HCLS   15      // HUMAN=14 -> bbox/prob column 15
#define IMG_H  50.0f
#define IMG_W  50.0f
#define CCH    512
#define FH     50
#define FW     50
#define FC_IN  25088   // 512*7*7
#define FC_OUT 4096
#define KSPLIT 16
#define KCH    (FC_IN / KSPLIT)   // 1568, divisible by 32 (49 WMMA steps)

// ---------------- Kernel 1: per-roi human-class bbox + clamp(+bug) + softmax + |area|
__global__ void k_prep(const float* __restrict__ scores, const float* __restrict__ off,
                       const float* __restrict__ rois, const float* __restrict__ scale,
                       float* __restrict__ bboxh, float* __restrict__ probh,
                       float* __restrict__ areaA, int N) {
    int j = blockIdx.x * blockDim.x + threadIdx.x;
    if (j >= N) return;
    float sc = scale[0];
    const float* r = rois + (size_t)j * 4;
    float r0 = r[0] / sc, r1 = r[1] / sc, r2 = r[2] / sc, r3 = r[3] / sc;
    float h = r2 - r0, w = r3 - r1;
    float cy = r0 + 0.5f * h, cx = r1 + 0.5f * w;
    const float* po = off + (size_t)j * (NCLS * 4) + HCLS * 4;  // flat cols 60..63
    float dy = po[0] * 0.1f, dx = po[1] * 0.1f, dh = po[2] * 0.2f, dw = po[3] * 0.2f;
    float ncy = dy * h + cy, ncx = dx * w + cx;
    float nh = expf(dh) * h, nw = expf(dw) * w;
    float y1 = ncy - 0.5f * nh, x1 = ncx - 0.5f * nw;
    float y2 = ncy + 0.5f * nh, x2 = ncx + 0.5f * nw;
    // faithful clamp incl. the (ar%3==1) bug: col60->[0,H], col61->[0,W], col62->[0,H], col63 unclamped
    y1 = fminf(fmaxf(y1, 0.f), IMG_H);
    x1 = fminf(fmaxf(x1, 0.f), IMG_W);
    y2 = fminf(fmaxf(y2, 0.f), IMG_H);
    float* bb = bboxh + (size_t)j * 4;
    bb[0] = y1; bb[1] = x1; bb[2] = y2; bb[3] = x2;
    areaA[j] = fabsf((y2 - y1) * (x2 - x1));
    // softmax column 15
    const float* s = scores + (size_t)j * NCLS;
    float mx = s[0];
    for (int k = 1; k < NCLS; ++k) mx = fmaxf(mx, s[k]);
    float sum = 0.f;
    for (int k = 0; k < NCLS; ++k) sum += expf(s[k] - mx);
    probh[j] = expf(s[HCLS] - mx) / sum;
}

// ---------------- Kernel 2: stable descending rank -> scatter order (matches jnp.argsort(-s))
__global__ void k_rank(const float* __restrict__ probh, int* __restrict__ order, int N) {
    int j = blockIdx.x * blockDim.x + threadIdx.x;
    if (j >= N) return;
    float sj = probh[j];
    int rank = 0;
    for (int i = 0; i < N; ++i) {
        float si = probh[i];
        rank += (si > sj) || (si == sj && i < j);
    }
    order[rank] = j;
}

// ---------------- Kernel 3: sequential greedy NMS (thresh 0) + max-|area| pick, single block
__global__ void k_nms(const int* __restrict__ order, const float* __restrict__ bboxh,
                      const float* __restrict__ probh, const float* __restrict__ areaA,
                      float* __restrict__ bo, float* __restrict__ as,
                      float* __restrict__ hdr, float* __restrict__ out, int N) {
    __shared__ int   s_supp[2048];
    __shared__ float s_box[4];
    __shared__ float s_area;
    __shared__ int   s_act;
    __shared__ float s_rv[256];
    __shared__ int   s_ri[256];
    const int tid = threadIdx.x, B = blockDim.x;
    // phase 0: gather sorted boxes + signed areas, clear suppression
    for (int j = tid; j < N; j += B) {
        int o = order[j];
        float b0 = bboxh[o * 4 + 0], b1 = bboxh[o * 4 + 1];
        float b2 = bboxh[o * 4 + 2], b3 = bboxh[o * 4 + 3];
        bo[j * 4 + 0] = b0; bo[j * 4 + 1] = b1; bo[j * 4 + 2] = b2; bo[j * 4 + 3] = b3;
        as[j] = (b2 - b0) * (b3 - b1);           // signed, like reference pairwise_iou
        s_supp[j] = 0;
    }
    __syncthreads();
    // phase 1: greedy scan
    for (int i = 0; i < N; ++i) {
        if (tid == 0) {
            s_act = (s_supp[i] == 0);
            if (s_act) {
                s_box[0] = bo[i * 4 + 0]; s_box[1] = bo[i * 4 + 1];
                s_box[2] = bo[i * 4 + 2]; s_box[3] = bo[i * 4 + 3];
                s_area = as[i];
            }
        }
        __syncthreads();
        if (s_act) {
            for (int j = i + 1 + tid; j < N; j += B) {
                if (!s_supp[j]) {
                    float ty = fmaxf(s_box[0], bo[j * 4 + 0]);
                    float tx = fmaxf(s_box[1], bo[j * 4 + 1]);
                    float by = fminf(s_box[2], bo[j * 4 + 2]);
                    float bx = fminf(s_box[3], bo[j * 4 + 3]);
                    float inter = fmaxf(by - ty, 0.f) * fmaxf(bx - tx, 0.f);
                    float iou = inter / (s_area + as[j] - inter + 1e-9f);
                    if (iou >= 0.f) s_supp[j] = 1;   // NaN -> false, like JAX
                }
            }
        }
        __syncthreads();
    }
    // phase 2: best = argmax over original index of (kept ? |area| : -1), first-max tie-break
    float bv = -3.0e38f; int bi = 0x7fffffff;
    for (int j = tid; j < N; j += B) {
        int o = order[j];
        float v = s_supp[j] ? -1.0f : areaA[o];
        if (v > bv || (v == bv && o < bi)) { bv = v; bi = o; }
    }
    s_rv[tid] = bv; s_ri[tid] = bi;
    __syncthreads();
    for (int s = B >> 1; s > 0; s >>= 1) {
        if (tid < s) {
            float v = s_rv[tid + s]; int i2 = s_ri[tid + s];
            if (v > s_rv[tid] || (v == s_rv[tid] && i2 < s_ri[tid])) { s_rv[tid] = v; s_ri[tid] = i2; }
        }
        __syncthreads();
    }
    if (tid == 0) {
        int best = s_ri[0];
        float b0 = bboxh[best * 4 + 0], b1 = bboxh[best * 4 + 1];
        float b2 = bboxh[best * 4 + 2], b3 = bboxh[best * 4 + 3];
        hdr[0] = b0; hdr[1] = b1; hdr[2] = b2; hdr[3] = b3;
        out[0] = b0; out[1] = b1; out[2] = b2; out[3] = b3;
        out[4] = probh[best];
    }
}

// ---------------- Kernel 4: Chainer-style RoI max pool -> f16 feature vector
__global__ void k_pool(const float* __restrict__ x, const float* __restrict__ hdr,
                       _Float16* __restrict__ poolh) {
    int c = blockIdx.x, t = threadIdx.x;
    if (t >= 49) return;
    int pi = t / 7, pj = t % 7;
    // box_xyxy = (hb[1], hb[0], hb[3], hb[2]); hdr is yxyx
    float y1 = hdr[0], x1 = hdr[1], y2 = hdr[2], x2 = hdr[3];
    float ymin = rintf(y1 * 0.0625f), xmin = rintf(x1 * 0.0625f);   // rintf = round half-to-even
    float ymax = rintf(y2 * 0.0625f), xmax = rintf(x2 * 0.0625f);
    float roi_h = fmaxf(ymax - ymin + 1.f, 1.f);
    float roi_w = fmaxf(xmax - xmin + 1.f, 1.f);
    float bh = roi_h / 7.f, bw = roi_w / 7.f;
    float hs = ymin + floorf((float)pi * bh), he = ymin + ceilf(((float)pi + 1.f) * bh);
    float ws_ = xmin + floorf((float)pj * bw), we = xmin + ceilf(((float)pj + 1.f) * bw);
    const float* f = x + (size_t)c * (FH * FW);
    float m = -1e30f;
    for (int r = 0; r < FH; ++r) {
        float rf = (float)r;
        if (rf >= hs && rf < he) {
            const float* row = f + r * FW;
            for (int cc = 0; cc < FW; ++cc) {
                float cf = (float)cc;
                if (cf >= ws_ && cf < we) m = fmaxf(m, row[cc]);
            }
        }
    }
    if (m < -5e29f) m = 0.f;   // empty-bin guard
    poolh[(size_t)c * 49 + pi * 7 + pj] = (_Float16)m;
}

// ---------------- Kernel 5: fc7 GEMV via v_wmma_f32_16x16x32_f16, K split 16x for MLP.
// 4096 waves total (16 K-chunks x 256 n-tiles); each writes a deterministic partial.
__global__ void k_fc1_wmma(const float* __restrict__ W1, const _Float16* __restrict__ ah,
                           float* __restrict__ part) {
    const int lane = threadIdx.x & 31;
    const int wid  = threadIdx.x >> 5;
    const int g    = blockIdx.x * (blockDim.x >> 5) + wid;
    const int nt   = g & 255;          // n-tile: 256 tiles of 16 outputs
    const int kc   = g >> 8;           // K-chunk: 0..15
    const int n    = nt * 16 + (lane & 15);
    const int kA   = (lane & 16) ? 8 : 0;    // A-frag K half-shift (lanes 16-31 hold K 8-15/24-31)
    const int kB   = (lane & 16) ? 8 : 0;    // B-frag K half-shift (lanes 16-31 hold K 8-15/24-31)
    const int kbeg = kc * KCH, kend = kbeg + KCH;
    v8f acc = {};
    for (int k0 = kbeg; k0 < kend; k0 += 32) {
        v16h A, B;
#pragma unroll
        for (int m = 0; m < 16; ++m) {
            // 16-bit A 16x32 layout: halfs 0..7 -> K m, halfs 8..15 -> K m+8 (then +8 for hi lanes)
            A[m] = ah[k0 + m + ((m & 8) ? 8 : 0) + kA];
        }
        const float* wp = W1 + (size_t)(k0 + kB) * FC_OUT + n;
#pragma unroll
        for (int m = 0; m < 16; ++m) {
            B[m] = (_Float16)wp[(size_t)(m + ((m & 8) ? 8 : 0)) * FC_OUT];  // B[k,n] column for this lane
        }
        acc = __builtin_amdgcn_wmma_f32_16x16x32_f16(
            /*neg_a=*/false, A, /*neg_b=*/false, B,
            /*c_mod=*/(short)0, acc, /*reuse_a=*/false, /*reuse_b=*/false);
        if (k0 + 32 < kend)
            __builtin_prefetch(wp + (size_t)32 * FC_OUT, 0, 1);   // global_prefetch next W1 tile
    }
    // all D rows identical (A rows replicated); acc[0] lanes 0-15 hold the 16 outputs
    if (lane < 16) part[(size_t)kc * FC_OUT + n] = acc[0];
}

// ---------------- Kernel 5b: deterministic fixed-order partial sum + bias + ReLU
__global__ void k_fc1_post(const float* __restrict__ part, const float* __restrict__ b1,
                           float* __restrict__ fc7) {
    int n = blockIdx.x * blockDim.x + threadIdx.x;
    if (n >= FC_OUT) return;
    float s = 0.f;
#pragma unroll
    for (int k = 0; k < KSPLIT; ++k) s += part[(size_t)k * FC_OUT + n];
    s += b1[n];
    fc7[n] = s > 0.f ? s : 0.f;
}

// ---------------- Kernel 6: heads (4096x4 and 4096x27), one wave32 per output
__global__ void k_heads(const float* __restrict__ fc7,
                        const float* __restrict__ Wl, const float* __restrict__ bl,
                        const float* __restrict__ Wa, const float* __restrict__ ba,
                        float* __restrict__ out) {
    const int wid = threadIdx.x >> 5, lane = threadIdx.x & 31;
    if (wid >= 31) return;
    const float* Wc; const float* bias; int ncol, col;
    if (wid < 4) { Wc = Wl; bias = bl; ncol = 4;  col = wid; }
    else         { Wc = Wa; bias = ba; ncol = 27; col = wid - 4; }
    float sum = 0.f;
    for (int k = lane; k < FC_OUT; k += 32)
        sum += fc7[k] * Wc[(size_t)k * ncol + col];
    for (int off = 16; off > 0; off >>= 1)
        sum += __shfl_down(sum, off, 32);
    if (lane == 0) out[5 + wid] = sum + bias[col];
}

extern "C" void kernel_launch(void* const* d_in, const int* in_sizes, int n_in,
                              void* d_out, int out_size, void* d_ws, size_t ws_size,
                              hipStream_t stream) {
    const float* x       = (const float*)d_in[0];
    const float* scores  = (const float*)d_in[1];
    const float* off     = (const float*)d_in[2];
    const float* rois    = (const float*)d_in[3];
    const float* W1      = (const float*)d_in[4];
    const float* b1      = (const float*)d_in[5];
    const float* Wl      = (const float*)d_in[6];
    const float* bl      = (const float*)d_in[7];
    const float* Wa      = (const float*)d_in[8];
    const float* ba      = (const float*)d_in[9];
    const float* scale   = (const float*)d_in[10];
    float* out = (float*)d_out;

    const int N = in_sizes[3] / 4;   // 2000 rois

    // workspace carve-up (256B aligned)
    char* base = (char*)d_ws;
    size_t o = 0;
    auto carve = [&](size_t bytes) { size_t r = o; o = (o + bytes + 255) & ~(size_t)255; return r; };
    float*     hdr   = (float*)(base + carve(64));
    float*     bboxh = (float*)(base + carve((size_t)N * 4 * sizeof(float)));
    float*     probh = (float*)(base + carve((size_t)N * sizeof(float)));
    float*     areaA = (float*)(base + carve((size_t)N * sizeof(float)));
    int*       order = (int*)  (base + carve((size_t)N * sizeof(int)));
    float*     bo    = (float*)(base + carve((size_t)N * 4 * sizeof(float)));
    float*     as    = (float*)(base + carve((size_t)N * sizeof(float)));
    _Float16*  poolh = (_Float16*)(base + carve((size_t)FC_IN * sizeof(_Float16)));
    float*     part  = (float*)(base + carve((size_t)KSPLIT * FC_OUT * sizeof(float)));
    float*     fc7   = (float*)(base + carve((size_t)FC_OUT * sizeof(float)));

    dim3 b256(256);
    k_prep<<<dim3((N + 255) / 256), b256, 0, stream>>>(scores, off, rois, scale,
                                                       bboxh, probh, areaA, N);
    k_rank<<<dim3((N + 255) / 256), b256, 0, stream>>>(probh, order, N);
    k_nms <<<dim3(1), b256, 0, stream>>>(order, bboxh, probh, areaA, bo, as, hdr, out, N);
    k_pool<<<dim3(CCH), dim3(64), 0, stream>>>(x, hdr, poolh);
    // 16 K-chunks x 256 n-tiles = 4096 waves = 512 blocks of 8 waves
    k_fc1_wmma<<<dim3(KSPLIT * 256 / 8), b256, 0, stream>>>(W1, poolh, part);
    k_fc1_post<<<dim3(FC_OUT / 256), b256, 0, stream>>>(part, b1, fc7);
    k_heads<<<dim3(1), dim3(1024), 0, stream>>>(fc7, Wl, bl, Wa, ba, out);
}
